// GraphInference_90202903151097
// MI455X (gfx1250) — compile-verified, matrix operand order checked by reading, not run on hardware
//
#include <hip/hip_runtime.h>
#include <hip/hip_bf16.h>

#define NN 12800     // nodes
#define NE 204800    // edges
#define DIM 16
#define MSG_WAVES 1600           // persistent waves in k_msg (200 blocks x 8)
#define GROUPS_PER_WAVE (NE/16/MSG_WAVES)   // 8

typedef _Float16 v16h __attribute__((ext_vector_type(16)));
typedef float    v8f  __attribute__((ext_vector_type(8)));

// ---------------- workspace layout (bytes) ----------------
#define WS_B      0                         // 9216 B   : f16 B operand (permuted nn2_w + bias rows)
#define WS_ROWPTR 16384                     // (NN+1)*4 : CSR row pointers
#define WS_CURSOR 69632                     // NN*4     : fill cursors
#define WS_DEGU   122880                    // NN*4     : integer in-degree
#define WS_PERM   176128                    // NE*4     : edge permutation (sorted by dst)
#define WS_MSG    995328                    // NE*16*4  : per-edge messages (sorted order)
#define WS_EH     (WS_MSG + NE*DIM*4)       // NE*16*2  : e_hidden (f16)

// out0 = relu(x @ lin0_w^T + lin0_b); zero integer degree
__global__ void k_node_init(const float* __restrict__ x,
                            const float* __restrict__ w,
                            const float* __restrict__ b,
                            float* __restrict__ out,
                            unsigned* __restrict__ degu) {
  int i = blockIdx.x * blockDim.x + threadIdx.x;
  if (i >= NN) return;
  float x0 = x[i*3+0], x1 = x[i*3+1], x2 = x[i*3+2];
#pragma unroll
  for (int k = 0; k < DIM; ++k) {
    float s = b[k] + x0*w[k*3+0] + x1*w[k*3+1] + x2*w[k*3+2];
    out[i*DIM+k] = s > 0.f ? s : 0.f;
  }
  degu[i] = 0u;
}

// e_hidden = relu(edge_attr @ nn1_w^T + nn1_b) stored f16; integer degree count
__global__ void k_edge_init(const float* __restrict__ attr,
                            const int*   __restrict__ dst,
                            const float* __restrict__ w1,
                            const float* __restrict__ b1,
                            _Float16* __restrict__ eh,
                            unsigned* __restrict__ degu) {
  int e = blockIdx.x * blockDim.x + threadIdx.x;
  if (e >= NE) return;
  float a0 = attr[e*4+0], a1 = attr[e*4+1], a2 = attr[e*4+2], a3 = attr[e*4+3];
#pragma unroll
  for (int k = 0; k < DIM; ++k) {
    float s = b1[k] + a0*w1[k*4+0] + a1*w1[k*4+1] + a2*w1[k*4+2] + a3*w1[k*4+3];
    eh[e*DIM+k] = (_Float16)(s > 0.f ? s : 0.f);
  }
  atomicAdd(&degu[dst[e]], 1u);
}

// Single-block LDS prefix scan: rowptr = exclusive_scan(deg); cursor = rowptr copy.
__global__ void k_scan(const unsigned* __restrict__ degu,
                       unsigned* __restrict__ rowptr,
                       unsigned* __restrict__ cursor) {
  __shared__ unsigned buf[256];
  __shared__ unsigned carry;
  if (threadIdx.x == 0) carry = 0u;
  __syncthreads();
  for (int base = 0; base < NN; base += 256) {
    int i = base + (int)threadIdx.x;
    unsigned v = (i < NN) ? degu[i] : 0u;
    buf[threadIdx.x] = v;
    __syncthreads();
#pragma unroll
    for (int off = 1; off < 256; off <<= 1) {
      unsigned t = (threadIdx.x >= (unsigned)off) ? buf[threadIdx.x - off] : 0u;
      __syncthreads();
      buf[threadIdx.x] += t;
      __syncthreads();
    }
    unsigned excl = carry + buf[threadIdx.x] - v;
    if (i < NN) { rowptr[i] = excl; cursor[i] = excl; }
    __syncthreads();
    if (threadIdx.x == 255) carry += buf[255];
    __syncthreads();
  }
  if (threadIdx.x == 0) rowptr[NN] = carry;   // == NE
}

// Scatter edge ids into CSR order (positions via int atomics).
__global__ void k_fill(const int* __restrict__ dst,
                       unsigned* __restrict__ cursor,
                       int* __restrict__ perm) {
  int e = blockIdx.x * blockDim.x + threadIdx.x;
  if (e >= NE) return;
  unsigned pos = atomicAdd(&cursor[dst[e]], 1u);
  perm[pos] = e;
}

// Canonicalize per-node edge order (insertion sort, ranges ~16) -> fully
// deterministic message placement and summation order.
__global__ void k_sort(const unsigned* __restrict__ rowptr,
                       int* __restrict__ perm) {
  int i = blockIdx.x * blockDim.x + threadIdx.x;
  if (i >= NN) return;
  unsigned r0 = rowptr[i], r1 = rowptr[i+1];
  for (unsigned a = r0 + 1; a < r1; ++a) {
    int key = perm[a];
    unsigned b = a;
    while (b > r0 && perm[b-1] > key) { perm[b] = perm[b-1]; --b; }
    perm[b] = key;
  }
}

// Pre-permute nn2_w (+nn2_b bias rows) into WMMA B-operand layout, f16.
// K=d*16+j (K<256): B[K,n]=nn2_w[d*16+n, j]; K in [256,272): B[K,n]=nn2_b[(K-256)*16+n];
// [272,288): zero pad.  Per-lane half t -> K: lanes<16 hold {0..7,16..23}, lanes>=16 rest.
__global__ void k_build_B(const float* __restrict__ nn2_w,
                          const float* __restrict__ nn2_b,
                          _Float16* __restrict__ B) {
  int c = blockIdx.x;          // chunk 0..8
  int lane = threadIdx.x;      // 0..31 ; N = lane & 15
  int n = lane & 15;
  int hi = (lane >= 16) ? 8 : 0;
#pragma unroll
  for (int t = 0; t < 16; ++t) {
    int K = c*32 + ((t & 8) << 1) + (t & 7) + hi;
    float v = 0.f;
    if (K < 256) {
      int d = K >> 4, j = K & 15;
      v = nn2_w[(d*16 + n)*16 + j];
    } else if (K < 272) {
      v = nn2_b[(K - 256)*16 + n];
    }
    B[(c*32 + lane)*16 + t] = (_Float16)v;
  }
}

// Fused message kernel: persistent waves, B held in 72 VGPRs across 8 groups.
// A[e, K=(d,j)] = a16[d] * h16[j] (+bias rows A=a16[d]) built with packed f16
// multiplies; msg(16x16) = A(16x288) @ B(288x16) via 9x v_wmma_f32_16x16x32_f16.
// Results stored (no atomics) at CSR positions into msg[].
__global__ void __launch_bounds__(256)
k_msg(const float* __restrict__ out,
      const _Float16* __restrict__ eh,
      const int*   __restrict__ src,
      const int*   __restrict__ perm,
      const _Float16* __restrict__ B,
      float* __restrict__ msg) {
  int lane = threadIdx.x & 31;
  int wave = threadIdx.x >> 5;
  int waveId = blockIdx.x * 8 + wave;          // 0..1599
  int hi = (lane >= 16) ? 8 : 0;
  int n = lane & 15;

  v16h Breg[9];
#pragma unroll
  for (int c = 0; c < 9; ++c)
    Breg[c] = *(const v16h*)(B + (c*32 + lane)*16);

  for (int g = 0; g < GROUPS_PER_WAVE; ++g) {
    int base = (waveId + g * MSG_WAVES) * 16;  // first CSR position of this group
    int ep = perm[base + n];                   // edge id at this position
    int s = src[ep];

    // h vector already f16 (one 32B load); a vector converted to f16 once.
    v16h hvv = *(const v16h*)(eh + ep * DIM);
    _Float16 av16[16];
    const float4* op = (const float4*)(out + s * DIM);
#pragma unroll
    for (int q = 0; q < 4; ++q) {
      float4 t0 = op[q];
      av16[q*4+0] = (_Float16)t0.x; av16[q*4+1] = (_Float16)t0.y;
      av16[q*4+2] = (_Float16)t0.z; av16[q*4+3] = (_Float16)t0.w;
    }

    v8f acc = {};
#pragma unroll
    for (int c = 0; c < 9; ++c) {
      v16h a;
#pragma unroll
      for (int t = 0; t < 16; ++t) {
        int K = c*32 + ((t & 8) << 1) + (t & 7) + hi;  // constant-folds per unroll
        _Float16 v;
        if (K < 256)      v = av16[K >> 4] * hvv[K & 15];   // packed f16 muls
        else if (K < 272) v = av16[K - 256];
        else              v = (_Float16)0.f;
        a[t] = v;
      }
      acc = __builtin_amdgcn_wmma_f32_16x16x32_f16(
          false, a, false, Breg[c], (short)0, acc, false, false);
    }

    // C tile: VGPR v -> rows v (lanes 0-15) and v+8 (lanes 16-31); col = lane&15
#pragma unroll
    for (int v = 0; v < 8; ++v)
      msg[(base + v + hi)*DIM + n] = acc[v];
  }
}

// Per-node: atomic-free segmented mean over CSR range, conv_root + relu, GRU step.
__global__ void k_node_update(float* __restrict__ hbuf,
                              const float* __restrict__ msg,
                              const unsigned* __restrict__ rowptr,
                              const float* __restrict__ cw,
                              const float* __restrict__ cb,
                              const float* __restrict__ wih,
                              const float* __restrict__ whh,
                              const float* __restrict__ bih,
                              const float* __restrict__ bhh) {
  int i = blockIdx.x * blockDim.x + threadIdx.x;
  if (i >= NN) return;
  unsigned r0 = rowptr[i], r1 = rowptr[i+1];
  float ag[16];
#pragma unroll
  for (int k = 0; k < 16; ++k) ag[k] = 0.f;
  for (unsigned r = r0; r < r1; ++r) {
    const float4* mp = (const float4*)(msg + r*DIM);
#pragma unroll
    for (int q = 0; q < 4; ++q) {
      float4 t = mp[q];
      ag[q*4+0] += t.x; ag[q*4+1] += t.y; ag[q*4+2] += t.z; ag[q*4+3] += t.w;
    }
  }
  float idg = (r1 > r0) ? 1.f / (float)(r1 - r0) : 1.f;

  float h[16], m[16];
#pragma unroll
  for (int k = 0; k < 16; ++k) h[k] = hbuf[i*DIM+k];
#pragma unroll
  for (int k = 0; k < 16; ++k) {
    float sacc = ag[k]*idg + cb[k];
#pragma unroll
    for (int d = 0; d < 16; ++d) sacc += h[d] * cw[d*16+k];
    m[k] = sacc > 0.f ? sacc : 0.f;
  }
#pragma unroll
  for (int k = 0; k < 16; ++k) {
    float gr  = bih[k]      + bhh[k];
    float gz  = bih[16 + k] + bhh[16 + k];
    float gin = bih[32 + k];
    float ghn = bhh[32 + k];
#pragma unroll
    for (int d = 0; d < 16; ++d) {
      gr  += m[d]*wih[k*16+d]      + h[d]*whh[k*16+d];
      gz  += m[d]*wih[(16+k)*16+d] + h[d]*whh[(16+k)*16+d];
      gin += m[d]*wih[(32+k)*16+d];
      ghn += h[d]*whh[(32+k)*16+d];
    }
    float r  = 1.f / (1.f + __expf(-gr));
    float z  = 1.f / (1.f + __expf(-gz));
    float nh = tanhf(gin + r*ghn);
    hbuf[i*DIM+k] = (1.f - z)*nh + z*h[k];
  }
}

extern "C" void kernel_launch(void* const* d_in, const int* in_sizes, int n_in,
                              void* d_out, int out_size, void* d_ws, size_t ws_size,
                              hipStream_t stream) {
  (void)in_sizes; (void)n_in; (void)out_size; (void)ws_size;
  const float* x      = (const float*)d_in[0];
  const int*   ei     = (const int*)  d_in[1];
  const float* eattr  = (const float*)d_in[2];
  const float* lin0_w = (const float*)d_in[3];
  const float* lin0_b = (const float*)d_in[4];
  const float* nn1_w  = (const float*)d_in[5];
  const float* nn1_b  = (const float*)d_in[6];
  const float* nn2_w  = (const float*)d_in[7];
  const float* nn2_b  = (const float*)d_in[8];
  const float* conv_w = (const float*)d_in[9];
  const float* conv_b = (const float*)d_in[10];
  const float* wih    = (const float*)d_in[11];
  const float* whh    = (const float*)d_in[12];
  const float* bih    = (const float*)d_in[13];
  const float* bhh    = (const float*)d_in[14];
  const int* src = ei;
  const int* dst = ei + NE;

  char* ws = (char*)d_ws;
  _Float16* B      = (_Float16*)(ws + WS_B);
  unsigned* rowptr = (unsigned*)(ws + WS_ROWPTR);
  unsigned* cursor = (unsigned*)(ws + WS_CURSOR);
  unsigned* degu   = (unsigned*)(ws + WS_DEGU);
  int*      perm   = (int*)     (ws + WS_PERM);
  float*    msg    = (float*)   (ws + WS_MSG);
  _Float16* eh     = (_Float16*)(ws + WS_EH);
  float*    hbuf   = (float*)d_out;   // holds out == h throughout

  k_node_init<<<(NN + 255)/256, 256, 0, stream>>>(x, lin0_w, lin0_b, hbuf, degu);
  k_edge_init<<<(NE + 255)/256, 256, 0, stream>>>(eattr, dst, nn1_w, nn1_b, eh, degu);
  k_scan<<<1, 256, 0, stream>>>(degu, rowptr, cursor);
  k_fill<<<(NE + 255)/256, 256, 0, stream>>>(dst, cursor, perm);
  k_sort<<<(NN + 255)/256, 256, 0, stream>>>(rowptr, perm);
  k_build_B<<<9, 32, 0, stream>>>(nn2_w, nn2_b, B);

  for (int r = 0; r < 6; ++r) {
    k_msg<<<MSG_WAVES/8, 256, 0, stream>>>(hbuf, eh, src, perm, B, msg);
    k_node_update<<<(NN + 255)/256, 256, 0, stream>>>(hbuf, msg, rowptr, conv_w, conv_b,
                                                      wih, whh, bih, bhh);
  }
}
